// OneSideInterModalityUpdate_86260123173699
// MI455X (gfx1250) — compile-verified, matrix operand order checked by reading, not run on hardware
//
#include <hip/hip_runtime.h>

// ---------- vector types ----------
typedef __attribute__((ext_vector_type(8)))  __bf16 v8bf;
typedef __attribute__((ext_vector_type(16))) __bf16 v16bf;
typedef __attribute__((ext_vector_type(8)))  float  v8f;

__device__ __forceinline__ v16bf load_frag16(const __bf16* p) {
  // 16-byte granular loads: every address in this file is 16B aligned,
  // but not necessarily 32B aligned (LDS row strides of 80/144 bytes).
  v8bf lo = *(const v8bf*)(p);
  v8bf hi = *(const v8bf*)(p + 8);
  v16bf r;
#pragma unroll
  for (int i = 0; i < 8; ++i) { r[i] = lo[i]; r[i + 8] = hi[i]; }
  return r;
}

__device__ __forceinline__ void store_frag16(__bf16* p, v16bf v) {
  v8bf lo, hi;
#pragma unroll
  for (int i = 0; i < 8; ++i) { lo[i] = v[i]; hi[i] = v[i + 8]; }
  *(v8bf*)(p)     = lo;
  *(v8bf*)(p + 8) = hi;
}

// ---------- problem constants ----------
#define BATCH   32
#define SEQ_S   1024
#define SEQ_T   512
#define NHEAD   8
#define DHEAD   128
#define OUTSZ   1024

// ============================================================
// 1) sum of squares reduction (for weight-norm ||v||_F)
// ============================================================
__global__ void sumsq_kernel(const float* __restrict__ v, long n,
                             float* __restrict__ out) {
  long i = (long)blockIdx.x * blockDim.x + threadIdx.x;
  long stride = (long)gridDim.x * blockDim.x;
  float s = 0.f;
  for (; i < n; i += stride) { float x = v[i]; s += x * x; }
#pragma unroll
  for (int off = 16; off > 0; off >>= 1) s += __shfl_xor(s, off, 32);
  if ((threadIdx.x & 31) == 0) atomicAdd(out, s);
}

// ============================================================
// 2) W_bf16 = bf16( v * g / ||v|| )
// ============================================================
__global__ void wn_to_bf16_kernel(const float* __restrict__ v,
                                  const float* __restrict__ g,
                                  const float* __restrict__ sumsq,
                                  __bf16* __restrict__ out, long n) {
  float scale = g[0] * rsqrtf(sumsq[0]);
  long i = (long)blockIdx.x * blockDim.x + threadIdx.x;
  long stride = (long)gridDim.x * blockDim.x;
  for (; i < n; i += stride) out[i] = (__bf16)(v[i] * scale);
}

// ============================================================
// 3) f32 -> bf16 with output leading dimension (for cat buffer)
// ============================================================
__global__ void cvt_bf16_kernel(const float* __restrict__ x,
                                __bf16* __restrict__ dst,
                                long rows, int cols, int ld) {
  long n = rows * cols;
  long i = (long)blockIdx.x * blockDim.x + threadIdx.x;
  long stride = (long)gridDim.x * blockDim.x;
  for (; i < n; i += stride) {
    long r = i / cols;
    int  c = (int)(i - r * cols);
    dst[r * (long)ld + c] = (__bf16)x[i];
  }
}

// ============================================================
// 4) Generic WMMA GEMM:  C[m,n] = sum_k A[m,k]*W[n,k] + bias[n]
//    Compile-time mask / output-type variants (no epilogue branches).
//    Block tile 128x128, 8 waves (4x2), wave tile 32x64,
//    K-step 32, register double-buffered global->LDS pipeline.
//    M,N multiples of 128; K multiple of 32 (true for all calls).
// ============================================================
#define BM  128
#define BN  128
#define BK  32
#define LDT 40   // LDS row stride in bf16 elements (80B, 16B aligned)

template <bool HAS_MASK, bool OUT_BF16>
__global__ __launch_bounds__(256) void gemm_wmma_kernel(
    const __bf16* __restrict__ A, int lda,
    const __bf16* __restrict__ W, int ldw,
    int K,
    const float* __restrict__ bias,
    const float* __restrict__ rowmask,   // used only when HAS_MASK
    void* __restrict__ outp,             // f32 or bf16 per OUT_BF16
    int ldo, int ocol) {
  __shared__ __bf16 As[BM * LDT];
  __shared__ __bf16 Bs[BN * LDT];

  const int tid  = threadIdx.x;
  const int wave = tid >> 5, lane = tid & 31;
  const int lcol = lane & 15, lhi = lane >> 4;
  const int wm = wave >> 1, wn = wave & 1;          // 4 x 2 wave grid
  const long m0 = (long)blockIdx.y * BM;
  const long n0 = (long)blockIdx.x * BN;

  const int ldrow  = tid >> 1;                      // 0..127
  const int ldhalf = (tid & 1) * 16;                // 0 or 16

  v8f acc[2][4];
  const v8f vzero = {0.f, 0.f, 0.f, 0.f, 0.f, 0.f, 0.f, 0.f};
#pragma unroll
  for (int mi = 0; mi < 2; ++mi)
#pragma unroll
    for (int ni = 0; ni < 4; ++ni) acc[mi][ni] = vzero;

  const __bf16* ag = A + (m0 + ldrow) * lda + ldhalf;
  const __bf16* wg = W + (n0 + ldrow) * ldw + ldhalf;
  __bf16* as_p = &As[ldrow * LDT + ldhalf];
  __bf16* bs_p = &Bs[ldrow * LDT + ldhalf];

  // register double-buffer: globals for chunk k issued while chunk k-1 computes
  v16bf areg = load_frag16(ag);
  v16bf wreg = load_frag16(wg);

  for (int k0 = 0; k0 < K; k0 += BK) {
    __syncthreads();                    // previous LDS reads complete
    store_frag16(as_p, areg);
    store_frag16(bs_p, wreg);
    __syncthreads();                    // tiles visible to all waves
    if (k0 + BK < K) {
      areg = load_frag16(ag + k0 + BK); // overlaps WMMA block below
      wreg = load_frag16(wg + k0 + BK);
      __builtin_prefetch(ag + k0 + 2 * BK, 0, 3);   // near-scope prefetch
      __builtin_prefetch(wg + k0 + 2 * BK, 0, 3);
    }

    v16bf af[2], bfg[4];
#pragma unroll
    for (int mi = 0; mi < 2; ++mi)
      af[mi] = load_frag16(&As[(wm * 32 + mi * 16 + lcol) * LDT + lhi * 16]);
#pragma unroll
    for (int ni = 0; ni < 4; ++ni)
      bfg[ni] = load_frag16(&Bs[(wn * 64 + ni * 16 + lcol) * LDT + lhi * 16]);

#pragma unroll
    for (int mi = 0; mi < 2; ++mi)
#pragma unroll
      for (int ni = 0; ni < 4; ++ni)
        acc[mi][ni] = __builtin_amdgcn_wmma_f32_16x16x32_bf16(
            false, af[mi], false, bfg[ni], (short)0, acc[mi][ni], false, false);
  }

  // ---- epilogue: hoisted bias / mask, branch-free ----
  float bvals[4];
#pragma unroll
  for (int ni = 0; ni < 4; ++ni)
    bvals[ni] = bias[n0 + wn * 64 + ni * 16 + lcol];

  float mvals[2][8];
  if (HAS_MASK) {
#pragma unroll
    for (int mi = 0; mi < 2; ++mi)
#pragma unroll
      for (int i = 0; i < 8; ++i)
        mvals[mi][i] = rowmask[m0 + wm * 32 + mi * 16 + i + 8 * lhi];
  }

#pragma unroll
  for (int mi = 0; mi < 2; ++mi) {
#pragma unroll
    for (int ni = 0; ni < 4; ++ni) {
#pragma unroll
      for (int i = 0; i < 8; ++i) {
        long m = m0 + wm * 32 + mi * 16 + i + 8 * lhi;
        long n = n0 + wn * 64 + ni * 16 + lcol;
        float vv = acc[mi][ni][i] + bvals[ni];
        if (HAS_MASK) vv *= mvals[mi][i];
        if (OUT_BF16)
          ((__bf16*)outp)[m * (long)ldo + ocol + n] = (__bf16)vv;
        else
          ((float*)outp)[m * (long)ldo + n] = vv;
      }
    }
  }
}

// ============================================================
// 5) Flash attention: per block one (b, h, 128-row T tile).
//    8 waves x 16 query rows. Online softmax over S in 64-chunks.
//    Q.K^T: both operands K-major (dh contiguous) -> direct frags.
//    P.V:   P restaged via LDS (C-layout -> A-layout),
//           V staged transposed in LDS so B-frags are contiguous.
//    Output written as bf16 into cols [1024,2048) of the cat buffer.
// ============================================================
__global__ __launch_bounds__(256) void flash_attn_kernel(
    const __bf16* __restrict__ q,     // [B*T, 1024] masked tgt projection
    const __bf16* __restrict__ kv,    // [B*S, 2048] masked src projection
    const float*  __restrict__ smask, // [B*S]
    __bf16* __restrict__ outcat) {    // [B*T, 2048]
  __shared__ __bf16 p_lds[8][16 * 64];        // per-wave P tile, row stride 64
  __shared__ __bf16 vt_lds[128 * 72];         // V^T [d][s], s padded to 72

  const int tid  = threadIdx.x;
  const int wave = tid >> 5, lane = tid & 31;
  const int lcol = lane & 15, lhi = lane >> 4;

  const int bid  = blockIdx.x;
  const int tblk = bid & 3;
  const int h    = (bid >> 2) & 7;
  const int b    = bid >> 5;
  const int trow0 = tblk * 128 + wave * 16;
  const float SCALE = 0.08838834764831845f;   // 1/sqrt(128)
  const float NEG   = -1e30f;

  // Q fragments for this wave's 16 rows, 4 chunks along dh
  v16bf qf[4];
  {
    const __bf16* qp = q + (long)(b * SEQ_T + trow0 + lcol) * OUTSZ
                         + h * DHEAD + lhi * 16;
#pragma unroll
    for (int kc = 0; kc < 4; ++kc) qf[kc] = load_frag16(qp + kc * 32);
  }

  v8f oacc[8];
  const v8f vzero = {0.f, 0.f, 0.f, 0.f, 0.f, 0.f, 0.f, 0.f};
#pragma unroll
  for (int oc = 0; oc < 8; ++oc) oacc[oc] = vzero;
  float mstat[8], lstat[8];
#pragma unroll
  for (int i = 0; i < 8; ++i) { mstat[i] = NEG; lstat[i] = 0.f; }

  for (int s0 = 0; s0 < SEQ_S; s0 += 64) {
    __syncthreads();                                  // protect vt_lds reuse
    {  // cooperative transposed stage of V chunk [64 s x 128 d]
      int sr = tid >> 2;                              // 0..63
      int dchunk = (tid & 3) * 32;
      const __bf16* vp = kv + (long)(b * SEQ_S + s0 + sr) * 2048
                            + OUTSZ + h * DHEAD + dchunk;
      v16bf v0 = load_frag16(vp);
      v16bf v1 = load_frag16(vp + 16);
#pragma unroll
      for (int j = 0; j < 16; ++j) {
        vt_lds[(dchunk + j)      * 72 + sr] = v0[j];
        vt_lds[(dchunk + 16 + j) * 72 + sr] = v1[j];
      }
    }
    float mk[4];
#pragma unroll
    for (int st = 0; st < 4; ++st)
      mk[st] = smask[b * SEQ_S + s0 + st * 16 + lcol];
    __syncthreads();

    // ---- scores: 16 WMMAs (4 s-subtiles x 4 dh chunks) ----
    v8f sacc[4];
#pragma unroll
    for (int st = 0; st < 4; ++st) sacc[st] = vzero;
    const __bf16* kbase = kv + (long)(b * SEQ_S + s0 + lcol) * 2048
                             + h * DHEAD + lhi * 16;
#pragma unroll
    for (int kc = 0; kc < 4; ++kc) {
#pragma unroll
      for (int st = 0; st < 4; ++st) {
        v16bf kf = load_frag16(kbase + (long)st * 16 * 2048 + kc * 32);
        sacc[st] = __builtin_amdgcn_wmma_f32_16x16x32_bf16(
            false, qf[kc], false, kf, (short)0, sacc[st], false, false);
      }
    }

    // ---- online softmax (rows live in vgpr slot i per 16-lane half) ----
    float nmax[8];
#pragma unroll
    for (int i = 0; i < 8; ++i) nmax[i] = mstat[i];
#pragma unroll
    for (int st = 0; st < 4; ++st)
#pragma unroll
      for (int i = 0; i < 8; ++i) {
        float v = sacc[st][i] * SCALE;
        v = (mk[st] == 0.f) ? NEG : v;
        sacc[st][i] = v;
        nmax[i] = fmaxf(nmax[i], v);
      }
#pragma unroll
    for (int i = 0; i < 8; ++i) {
#pragma unroll
      for (int off = 1; off < 16; off <<= 1)
        nmax[i] = fmaxf(nmax[i], __shfl_xor(nmax[i], off, 32));
    }
    float corr[8], rsum[8];
#pragma unroll
    for (int i = 0; i < 8; ++i) {
      corr[i] = __expf(mstat[i] - nmax[i]);
      mstat[i] = nmax[i];
      rsum[i] = 0.f;
    }
#pragma unroll
    for (int st = 0; st < 4; ++st)
#pragma unroll
      for (int i = 0; i < 8; ++i) {
        float p = __expf(sacc[st][i] - nmax[i]);
        rsum[i] += p;
        p_lds[wave][(i + 8 * lhi) * 64 + st * 16 + lcol] = (__bf16)p;
      }
#pragma unroll
    for (int i = 0; i < 8; ++i) {
#pragma unroll
      for (int off = 1; off < 16; off <<= 1)
        rsum[i] += __shfl_xor(rsum[i], off, 32);
      lstat[i] = lstat[i] * corr[i] + rsum[i];
    }
#pragma unroll
    for (int oc = 0; oc < 8; ++oc)
#pragma unroll
      for (int i = 0; i < 8; ++i) oacc[oc][i] *= corr[i];

    // ---- O += P @ V : 16 WMMAs (2 s-halves x 8 dh tiles) ----
#pragma unroll
    for (int kc2 = 0; kc2 < 2; ++kc2) {
      v16bf af = load_frag16(&p_lds[wave][lcol * 64 + kc2 * 32 + lhi * 16]);
#pragma unroll
      for (int oc = 0; oc < 8; ++oc) {
        v16bf vf = load_frag16(&vt_lds[(oc * 16 + lcol) * 72 + kc2 * 32 + lhi * 16]);
        oacc[oc] = __builtin_amdgcn_wmma_f32_16x16x32_bf16(
            false, af, false, vf, (short)0, oacc[oc], false, false);
      }
    }
  }

  // ---- write normalized O into cat buffer cols [1024, 2048) ----
#pragma unroll
  for (int i = 0; i < 8; ++i) {
    float inv = 1.0f / lstat[i];
    long row = (long)(b * SEQ_T + trow0 + i + 8 * lhi);
#pragma unroll
    for (int oc = 0; oc < 8; ++oc)
      outcat[row * 2048 + OUTSZ + h * DHEAD + oc * 16 + lcol] =
          (__bf16)(oacc[oc][i] * inv);
  }
}

// ============================================================
// launcher
// ============================================================
extern "C" void kernel_launch(void* const* d_in, const int* in_sizes, int n_in,
                              void* d_out, int out_size, void* d_ws, size_t ws_size,
                              hipStream_t stream) {
  (void)in_sizes; (void)n_in; (void)out_size; (void)ws_size;
  const float* src      = (const float*)d_in[0];
  const float* tgt      = (const float*)d_in[1];
  const float* src_mask = (const float*)d_in[2];
  const float* tgt_mask = (const float*)d_in[3];
  const float* v_src    = (const float*)d_in[4];
  const float* g_src    = (const float*)d_in[5];
  const float* b_src    = (const float*)d_in[6];
  const float* v_tgt    = (const float*)d_in[7];
  const float* g_tgt    = (const float*)d_in[8];
  const float* b_tgt    = (const float*)d_in[9];
  const float* v_out    = (const float*)d_in[10];
  const float* g_out    = (const float*)d_in[11];
  const float* b_out    = (const float*)d_in[12];

  char* ws = (char*)d_ws;
  size_t off = 0;
  auto carve = [&](size_t bytes) -> char* {
    char* p = ws + off;
    off = (off + bytes + 255) & ~((size_t)255);
    return p;
  };
  float*  norms = (float*) carve(16);                           // 3 sumsq slots
  __bf16* wsrc  = (__bf16*)carve((size_t)2048 * 1024 * 2);      // W_src  [2048,1024]
  __bf16* wtgt  = (__bf16*)carve((size_t)1024 * 1024 * 2);      // W_tgt  [1024,1024]
  __bf16* wout  = (__bf16*)carve((size_t)1024 * 2048 * 2);      // W_out  [1024,2048]
  __bf16* srcb  = (__bf16*)carve((size_t)32768 * 1024 * 2);     // bf16(src)
  __bf16* kvb   = (__bf16*)carve((size_t)32768 * 2048 * 2);     // masked src proj
  __bf16* catb  = (__bf16*)carve((size_t)16384 * 2048 * 2);     // [tgt | tgt_update]
  __bf16* qb    = (__bf16*)carve((size_t)16384 * 1024 * 2);     // masked tgt proj

  (void)hipMemsetAsync(norms, 0, 16, stream);
  sumsq_kernel<<<512, 256, 0, stream>>>(v_src, (long)2048 * 1024, norms + 0);
  sumsq_kernel<<<512, 256, 0, stream>>>(v_tgt, (long)1024 * 1024, norms + 1);
  sumsq_kernel<<<512, 256, 0, stream>>>(v_out, (long)1024 * 2048, norms + 2);

  wn_to_bf16_kernel<<<2048, 256, 0, stream>>>(v_src, g_src, norms + 0, wsrc, (long)2048 * 1024);
  wn_to_bf16_kernel<<<2048, 256, 0, stream>>>(v_tgt, g_tgt, norms + 1, wtgt, (long)1024 * 1024);
  wn_to_bf16_kernel<<<2048, 256, 0, stream>>>(v_out, g_out, norms + 2, wout, (long)1024 * 2048);

  cvt_bf16_kernel<<<4096, 256, 0, stream>>>(src, srcb, (long)32768, 1024, 1024);
  cvt_bf16_kernel<<<4096, 256, 0, stream>>>(tgt, catb, (long)16384, 1024, 2048);

  // src projection: [32768,1024] x [2048,1024]^T -> masked bf16 K|V
  dim3 g1(2048 / BN, 32768 / BM);
  gemm_wmma_kernel<true, true><<<g1, 256, 0, stream>>>(
      srcb, 1024, wsrc, 1024, 1024, b_src, src_mask, kvb, 2048, 0);
  // tgt projection: [16384,1024](lda=2048 in cat) x [1024,1024]^T -> masked bf16 Q
  dim3 g2(1024 / BN, 16384 / BM);
  gemm_wmma_kernel<true, true><<<g2, 256, 0, stream>>>(
      catb, 2048, wtgt, 1024, 1024, b_tgt, tgt_mask, qb, 1024, 0);
  // attention -> cat cols [1024,2048)
  flash_attn_kernel<<<BATCH * NHEAD * (SEQ_T / 128), 256, 0, stream>>>(
      qb, kvb, src_mask, catb);
  // output projection: [16384,2048] x [1024,2048]^T -> f32 d_out
  dim3 g3(1024 / BN, 16384 / BM);
  gemm_wmma_kernel<false, false><<<g3, 256, 0, stream>>>(
      catb, 2048, wout, 2048, 2048, b_out, nullptr, (float*)d_out, 1024, 0);
}
